// MaxCDFdp_multiclass_14439680049217
// MI455X (gfx1250) — matrix operand order repaired and checked by reading, not compiled, not run on hardware
//
#include <hip/hip_runtime.h>

#define N_SAMPLES 50000
#define N_CLASSES 20
#define N_PROBES  100
#define P_PAD     112          // 7 WMMA tiles of 16 probes
#define HTEMP     5.0f         // TEMPERATURE / 2 (tanh half-angle form)
#define CHUNK     2048
#define LOG2E     1.4426950408889634f

typedef __attribute__((ext_vector_type(16))) _Float16 v16h;
typedef __attribute__((ext_vector_type(8)))  float    v8f;

// tanh: prefer the gfx1250 v_tanh_f32 trans op; fall back to exp2+rcp identity.
#if __has_builtin(__builtin_amdgcn_tanhf)
__device__ __forceinline__ float dev_tanh(float x) { return __builtin_amdgcn_tanhf(x); }
#elif __has_builtin(__builtin_amdgcn_tanh_f32)
__device__ __forceinline__ float dev_tanh(float x) { return __builtin_amdgcn_tanh_f32(x); }
#else
__device__ __forceinline__ float dev_tanh(float x) {
  // tanh(x) = 2*sigmoid(2x) - 1 = 2/(1+exp2(-2*log2e*x)) - 1
  float e = __builtin_amdgcn_exp2f(x * (-2.0f * LOG2E));
  return fmaf(2.0f, __builtin_amdgcn_rcpf(1.0f + e), -1.0f);
}
#endif

// workspace layout (floats):
//  [0..19] mn[c], [20..39] mx[c], [40] n1 (count of s==1)
//  [WS_S0 + c*P_PAD + p] = S0 = sum_n tanh(...)*mask0,  same for WS_S1
#define WS_MN   0
#define WS_MX   20
#define WS_N1   40
#define WS_S0   64
#define WS_S1   (WS_S0 + N_CLASSES * P_PAD)
#define WS_TOTAL (WS_S0 + 2 * N_CLASSES * P_PAD)

// ---------------- Kernel 1: per-class min/max (+ group-1 count) ----------------
__global__ void minmax_kernel(const float* __restrict__ y,
                              const int* __restrict__ s,
                              float* __restrict__ ws) {
  __shared__ float smn[256], smx[256];
  __shared__ int   scnt[256];
  const int c = blockIdx.x, t = threadIdx.x;
  float mn = 3.4e38f, mx = -3.4e38f; int cnt = 0;
  for (int n = t; n < N_SAMPLES; n += 256) {
    float v = y[n * N_CLASSES + c];
    mn = fminf(mn, v); mx = fmaxf(mx, v);
    if (c == 0) cnt += s[n];
  }
  smn[t] = mn; smx[t] = mx; scnt[t] = cnt;
  __syncthreads();
  for (int off = 128; off > 0; off >>= 1) {
    if (t < off) {
      smn[t] = fminf(smn[t], smn[t + off]);
      smx[t] = fmaxf(smx[t], smx[t + off]);
      scnt[t] += scnt[t + off];
    }
    __syncthreads();
  }
  if (t == 0) {
    ws[WS_MN + c] = smn[0];
    ws[WS_MX + c] = smx[0];
    if (c == 0) ws[WS_N1] = (float)scnt[0];
  }
}

// ---------------- Kernel 2: tanh tiles + WMMA masked reduction ----------------
// grid = (ceil(N/CHUNK), C), block = 224 threads = 7 waves, wave w owns probes [16w,16w+16)
__global__ void __launch_bounds__(224)
cdf_kernel(const float* __restrict__ y,
           const int* __restrict__ s,
           float* __restrict__ ws) {
  __shared__ __align__(16) float    s_hy[CHUNK];   // (T/2)*y[n,c]
  __shared__ __align__(32) _Float16 s_m0[CHUNK];   // (s==0) ? 1 : 0, f16
  __shared__ __align__(32) _Float16 s_m1[CHUNK];   // (s==1) ? 1 : 0, f16
  __shared__ float s_hg[P_PAD];                    // (T/2)*grid[p]

  const int c    = blockIdx.y;
  const int base = blockIdx.x * CHUNK;
  const int tid  = threadIdx.x;

  const float mnc = ws[WS_MN + c], mxc = ws[WS_MX + c];
  for (int p = tid; p < P_PAD; p += blockDim.x) {
    int pp = (p < N_PROBES) ? p : (N_PROBES - 1);
    s_hg[p] = HTEMP * (mnc + (mxc - mnc) * ((float)pp / (float)(N_PROBES - 1)));
  }
  for (int i = tid; i < CHUNK; i += blockDim.x) {
    int n = base + i;
    if (n < N_SAMPLES) {
      s_hy[i] = HTEMP * y[n * N_CLASSES + c];
      int sv  = s[n];
      s_m0[i] = (sv == 0) ? (_Float16)1.0f : (_Float16)0.0f;
      s_m1[i] = (sv == 0) ? (_Float16)0.0f : (_Float16)1.0f;
    } else {                       // padding tail: zero masks -> zero contribution
      s_hy[i] = 0.0f;
      s_m0[i] = (_Float16)0.0f;
      s_m1[i] = (_Float16)0.0f;
    }
  }
  __syncthreads();

  const int lane = tid & 31;
  const int wv   = tid >> 5;        // 0..6 probe tile
  const int pb   = wv * 16;
  const int row  = lane & 15;       // A row (M) == B column (N)
  const int hi   = (lane >> 4) & 1;
  const float hg_row = s_hg[pb + row];

  // B-column base pointer: col0 -> mask0, col1 -> mask1, cols>=2 -> m0 (those
  // C columns are never read back, finite garbage is harmless).
  const _Float16* bp = (row == 1) ? s_m1 : s_m0;

  v8f acc = {};
  for (int k = 0; k < CHUNK; k += 32) {
    // 16-bit A 16x32 layout: half j -> K = j + 8*(lane>=16) + 8*(j>=8)
    const int a_lo = k + (hi ? 8 : 0);        // j = 0..7
    const int a_hi = k + 16 + (hi ? 8 : 0);   // j = 8..15
    const float4 f0 = *(const float4*)&s_hy[a_lo];
    const float4 f1 = *(const float4*)&s_hy[a_lo + 4];
    const float4 f2 = *(const float4*)&s_hy[a_hi];
    const float4 f3 = *(const float4*)&s_hy[a_hi + 4];
    v16h a;
    a[0]  = (_Float16)dev_tanh(hg_row - f0.x);
    a[1]  = (_Float16)dev_tanh(hg_row - f0.y);
    a[2]  = (_Float16)dev_tanh(hg_row - f0.z);
    a[3]  = (_Float16)dev_tanh(hg_row - f0.w);
    a[4]  = (_Float16)dev_tanh(hg_row - f1.x);
    a[5]  = (_Float16)dev_tanh(hg_row - f1.y);
    a[6]  = (_Float16)dev_tanh(hg_row - f1.z);
    a[7]  = (_Float16)dev_tanh(hg_row - f1.w);
    a[8]  = (_Float16)dev_tanh(hg_row - f2.x);
    a[9]  = (_Float16)dev_tanh(hg_row - f2.y);
    a[10] = (_Float16)dev_tanh(hg_row - f2.z);
    a[11] = (_Float16)dev_tanh(hg_row - f2.w);
    a[12] = (_Float16)dev_tanh(hg_row - f3.x);
    a[13] = (_Float16)dev_tanh(hg_row - f3.y);
    a[14] = (_Float16)dev_tanh(hg_row - f3.z);
    a[15] = (_Float16)dev_tanh(hg_row - f3.w);

    // 16-bit B 32x16 layout: half j -> K = j + 8*(lane>=16) + 8*(j>=8)
    const int kb = k + (hi ? 8 : 0);
    typedef __attribute__((ext_vector_type(8))) _Float16 v8h_t;
    const v8h_t b0 = *(const v8h_t*)&bp[kb];
    const v8h_t b1 = *(const v8h_t*)&bp[kb + 16];
    v16h b = __builtin_shufflevector(b0, b1, 0,1,2,3,4,5,6,7,8,9,10,11,12,13,14,15);

    acc = __builtin_amdgcn_wmma_f32_16x16x32_f16(
        /*neg_a=*/false, a, /*neg_b=*/false, b,
        /*c_mod=*/(short)0, acc, /*reuse_a=*/false, /*reuse_b=*/false);
  }

  // C/D layout: lane<16 -> (M=r, N=lane); lane>=16 -> (M=8+r, N=lane-16)
  const int col = lane & 15;
  if (col < 2) {
    float* dst = ws + (col == 0 ? WS_S0 : WS_S1) + c * P_PAD + pb + hi * 8;
#pragma unroll
    for (int r = 0; r < 8; ++r) atomicAdd(&dst[r], acc[r]);
  }
}

// ---------------- Kernel 3: delta + global max ----------------
// cdf_g = (0.5*S_g + 0.5*n_g)/n_g  =>  |cdf0-cdf1| = 0.5*|S0/n0 - S1/n1|
__global__ void finalize_kernel(const float* __restrict__ ws, float* __restrict__ out) {
  __shared__ float sm[256];
  const int t = threadIdx.x;
  const float n1 = ws[WS_N1];
  const float r0 = 1.0f / ((float)N_SAMPLES - n1);
  const float r1 = 1.0f / n1;
  float m = 0.0f;
  for (int idx = t; idx < N_CLASSES * N_PROBES; idx += 256) {
    int cc = idx / N_PROBES, p = idx % N_PROBES;
    float d = 0.5f * fabsf(ws[WS_S0 + cc * P_PAD + p] * r0 -
                           ws[WS_S1 + cc * P_PAD + p] * r1);
    m = fmaxf(m, d);
  }
  sm[t] = m;
  __syncthreads();
  for (int off = 128; off > 0; off >>= 1) {
    if (t < off) sm[t] = fmaxf(sm[t], sm[t + off]);
    __syncthreads();
  }
  if (t == 0) out[0] = sm[0];
}

extern "C" void kernel_launch(void* const* d_in, const int* in_sizes, int n_in,
                              void* d_out, int out_size, void* d_ws, size_t ws_size,
                              hipStream_t stream) {
  const float* y = (const float*)d_in[0];
  const int*   s = (const int*)d_in[1];
  float* ws  = (float*)d_ws;
  float* out = (float*)d_out;

  hipMemsetAsync(d_ws, 0, WS_TOTAL * sizeof(float), stream);
  minmax_kernel<<<N_CLASSES, 256, 0, stream>>>(y, s, ws);
  dim3 grid((N_SAMPLES + CHUNK - 1) / CHUNK, N_CLASSES);
  cdf_kernel<<<grid, 224, 0, stream>>>(y, s, ws);
  finalize_kernel<<<1, 256, 0, stream>>>(ws, out);
}
